// CausalSelfAttention_77386720739527
// MI455X (gfx1250) — compile-verified, matrix operand order checked by reading, not run on hardware
//
#include <hip/hip_runtime.h>

// ---------------------------------------------------------------------------
// CDNA5 (gfx1250) causal self-attention, bf16 WMMA pipeline, wave32.
// v4: GEMM stages its 16x1024 A panel in LDS (shared by 8 waves, async DMA);
//     flash attention double-buffers async K/V staging (1 barrier / step).
// ---------------------------------------------------------------------------

typedef __bf16 bf16;
typedef __attribute__((ext_vector_type(16))) __bf16 v16bf;
typedef __attribute__((ext_vector_type(8)))  __bf16 v8bf;
typedef __attribute__((ext_vector_type(8)))  float  v8f;
typedef __attribute__((ext_vector_type(4)))  int    v4i;

#define TT 4096
#define CC 1024
#define HH 16
#define DK 64
#define C3 3072

// ---- async global->LDS copy (ASYNCcnt path), with sync fallback -----------

#if defined(__gfx1250__) && __has_builtin(__builtin_amdgcn_global_load_async_to_lds_b128)
#define HAVE_ASYNC_LDS 1
typedef __attribute__((address_space(1))) v4i v4i_g;   // global int4
typedef __attribute__((address_space(3))) v4i v4i_l;   // LDS int4
#endif

__device__ __forceinline__ void copy16_to_lds(const bf16* g, bf16* l) {
#if defined(HAVE_ASYNC_LDS)
  __builtin_amdgcn_global_load_async_to_lds_b128(
      (v4i_g*)(unsigned long long)(uintptr_t)g,
      (v4i_l*)(unsigned int)(uintptr_t)l, 0, 0);
#else
  *reinterpret_cast<v8bf*>(l) = *reinterpret_cast<const v8bf*>(g);
#endif
}

__device__ __forceinline__ void copy_wait() {
#if defined(HAVE_ASYNC_LDS)
#if __has_builtin(__builtin_amdgcn_s_wait_asynccnt)
  __builtin_amdgcn_s_wait_asynccnt(0);
#else
  asm volatile("s_wait_asynccnt 0x0" ::: "memory");
#endif
#endif
}

// ---- WMMA helpers ---------------------------------------------------------

__device__ __forceinline__ v8f wmma_bf16(v16bf a, v16bf b, v8f c) {
  return __builtin_amdgcn_wmma_f32_16x16x32_bf16(
      /*neg_a=*/false, a, /*neg_b=*/false, b,
      /*c_mod=*/(short)0, c, /*reuse_a=*/false, /*reuse_b=*/false);
}

// 16x32 bf16 operand fragment in the A/B VGPR layout.  base -> (row0, k0) of
// row-major [rows][stride]; lane L: row = L&15, kb = 8*(L>=16);
// a[0..7] = K(kb..kb+7), a[8..15] = K(16+kb..23+kb).
__device__ __forceinline__ v16bf load_frag(const bf16* base, int stride, int kcol) {
  const int lane = threadIdx.x & 31;
  const bf16* p = base + (size_t)(lane & 15) * stride + kcol + ((lane >> 4) << 3);
  v8bf lo = *reinterpret_cast<const v8bf*>(p);
  v8bf hi = *reinterpret_cast<const v8bf*>(p + 16);
  v16bf a;
#pragma unroll
  for (int i = 0; i < 8; ++i) { a[i] = lo[i]; a[i + 8] = hi[i]; }
  return a;
}

// Row-wise reductions across the 16-lane half holding one C/D row.
__device__ __forceinline__ float half_reduce_max(float v) {
#pragma unroll
  for (int m = 1; m <= 8; m <<= 1) v = fmaxf(v, __shfl_xor(v, m, 32));
  return v;
}
__device__ __forceinline__ float half_reduce_sum(float v) {
#pragma unroll
  for (int m = 1; m <= 8; m <<= 1) v += __shfl_xor(v, m, 32);
  return v;
}

// ---- precision / layout conversion kernels --------------------------------

__global__ void k_cvt(const float* __restrict__ in, bf16* __restrict__ out, int n) {
  int i = blockIdx.x * blockDim.x + threadIdx.x;
  if (i < n) out[i] = (bf16)in[i];
}

// W [K][N] f32 row-major -> Wt [N][K] bf16
__global__ void k_cvt_T(const float* __restrict__ W, bf16* __restrict__ Wt,
                        int K, int N) {
  int i = blockIdx.x * blockDim.x + threadIdx.x;
  if (i < K * N) {
    int k = i / N, n = i - k * N;
    Wt[(size_t)n * K + k] = (bf16)W[i];
  }
}

// ---- WMMA GEMM: block = 16x512 tile, 8 waves x (16x64).  The 16xK A panel
// is staged once into LDS via async DMA and shared by all 8 waves.
// C[M][N] = A[M][K] * Bt[N][K]^T    (requires K <= 1024, K % 32 == 0)
template<bool OUT_BF16>
__global__ void __launch_bounds__(256) k_gemm(const bf16* __restrict__ A,
                                              const bf16* __restrict__ Bt,
                                              void* __restrict__ Cout,
                                              int M, int N, int K) {
  __shared__ __align__(16) bf16 As[16][1032];    // 16 x K panel (padded)
  const int tid   = threadIdx.x;
  const int wslot = tid >> 5;
  const int panels_n = N >> 9;                   // 512 columns per block
  const int tm = blockIdx.x / panels_n;
  const int n0 = (blockIdx.x % panels_n) * 512 + (wslot << 6);
  const bf16* Ab = A + (size_t)tm * 16 * K;

  const int kchunks = K >> 3;                    // 16-byte chunks per row
  for (int c = tid; c < 16 * kchunks; c += 256) {
    int row = c / kchunks, col = (c - row * kchunks) << 3;
    copy16_to_lds(&Ab[(size_t)row * K + col], &As[row][col]);
  }
  copy_wait();
  __syncthreads();

  const bf16* Bb = Bt + (size_t)n0 * K;
  v8f acc0 = {}, acc1 = {}, acc2 = {}, acc3 = {};
#pragma unroll 2
  for (int k = 0; k < K; k += 32) {
    v16bf a = load_frag(&As[0][0], 1032, k);     // shared A fragment (LDS)
    acc0 = wmma_bf16(a, load_frag(Bb + (size_t) 0 * K, K, k), acc0);
    acc1 = wmma_bf16(a, load_frag(Bb + (size_t)16 * K, K, k), acc1);
    acc2 = wmma_bf16(a, load_frag(Bb + (size_t)32 * K, K, k), acc2);
    acc3 = wmma_bf16(a, load_frag(Bb + (size_t)48 * K, K, k), acc3);
  }
  const int lane = threadIdx.x & 31;
  const int n  = n0 + (lane & 15);
  const int m0 = (tm << 4) + ((lane >> 4) << 3);
  if (OUT_BF16) {
    bf16* C = (bf16*)Cout;
#pragma unroll
    for (int r = 0; r < 8; ++r) {
      size_t row = (size_t)(m0 + r) * N + n;
      C[row +  0] = (bf16)acc0[r];
      C[row + 16] = (bf16)acc1[r];
      C[row + 32] = (bf16)acc2[r];
      C[row + 48] = (bf16)acc3[r];
    }
  } else {
    float* C = (float*)Cout;
#pragma unroll
    for (int r = 0; r < 8; ++r) {
      size_t row = (size_t)(m0 + r) * N + n;
      C[row +  0] = acc0[r];
      C[row + 16] = acc1[r];
      C[row + 32] = acc2[r];
      C[row + 48] = acc3[r];
    }
  }
}

// ---- RoPE + head split; V stored transposed [H][DK][T] --------------------

__global__ void k_rope(const bf16* __restrict__ qkv,
                       const float* __restrict__ fc, const float* __restrict__ fs,
                       bf16* __restrict__ Q, bf16* __restrict__ Kh,
                       bf16* __restrict__ Vt) {
  int tid = blockIdx.x * blockDim.x + threadIdx.x;   // T*H*(DK/2)
  int t = tid >> 9;
  int h = (tid >> 5) & 15;
  int i = tid & 31;
  float c = fc[t * 32 + i], s = fs[t * 32 + i];
  size_t base = (size_t)t * C3 + h * DK + 2 * i;
  size_t qo   = ((size_t)h * TT + t) * DK + 2 * i;
  float a0 = (float)qkv[base],      a1 = (float)qkv[base + 1];
  Q[qo]     = (bf16)(a0 * c - a1 * s);
  Q[qo + 1] = (bf16)(a0 * s + a1 * c);
  float b0 = (float)qkv[base + CC], b1 = (float)qkv[base + CC + 1];
  Kh[qo]     = (bf16)(b0 * c - b1 * s);
  Kh[qo + 1] = (bf16)(b0 * s + b1 * c);
  Vt[(size_t)(h * DK + 2 * i)     * TT + t] = qkv[base + 2 * CC];
  Vt[(size_t)(h * DK + 2 * i + 1) * TT + t] = qkv[base + 2 * CC + 1];
}

// ---- block-cooperative flash attention, double-buffered async staging -----
// block = 8 waves = 128 queries of one head.  K/V tiles for step i+1 are
// DMA'd into the alternate LDS buffer while step i computes; a single
// barrier per step is sufficient (a wave passes barrier(i) only after
// finishing compute(i-1), so writes to buf^1 after barrier(i) cannot race).

__global__ void __launch_bounds__(256) k_attn(const bf16* __restrict__ Q,
                                              const bf16* __restrict__ Kh,
                                              const bf16* __restrict__ Vt,
                                              bf16* __restrict__ yb) {
  __shared__ __align__(16) bf16 Ks[2][32][72];   // 32 keys x 64 d (padded)
  __shared__ __align__(16) bf16 Vs[2][64][40];   // 64 d x 32 keys (padded)
  __shared__ __align__(16) bf16 Ps[8][16][40];   // per-wave P tile

  const int tid   = threadIdx.x;
  const int wslot = tid >> 5;
  const int lane  = tid & 31;
  const int h     = blockIdx.x >> 5;             // 32 q-blocks per head
  const int qb0   = (blockIdx.x & 31) << 7;      // 128 queries per block
  const int q0    = qb0 + (wslot << 4);          // this wave's query tile
  const int nlane = lane & 15;
  const int mbase = (lane >> 4) << 3;
  bf16 (*sl)[40] = Ps[wslot];

  auto stage = [&](int buf, int k0) {
    const int kr = tid >> 3, kc = (tid & 7) << 3;        // 32x8 chunks
    copy16_to_lds(&Kh[((size_t)h * TT + k0 + kr) * DK + kc], &Ks[buf][kr][kc]);
    const int vr = tid >> 2, vc = (tid & 3) << 3;        // 64x4 chunks
    copy16_to_lds(&Vt[((size_t)h * DK + vr) * TT + k0 + vc], &Vs[buf][vr][vc]);
  };

  const bf16* Qb = Q + ((size_t)h * TT + q0) * DK;
  v16bf qa0 = load_frag(Qb, DK, 0);
  v16bf qa1 = load_frag(Qb, DK, 32);

  v8f o0 = {}, o1 = {}, o2 = {}, o3 = {};
  float row_max[8], row_sum[8];
#pragma unroll
  for (int r = 0; r < 8; ++r) { row_max[r] = -__builtin_inff(); row_sum[r] = 0.0f; }

  const int kmax = qb0 + 128;                    // block-uniform trip count
  stage(0, 0);                                   // prologue prefetch
  for (int k0 = 0; k0 < kmax; k0 += 32) {
    const int buf = (k0 >> 5) & 1;
    copy_wait();                                 // tile k0 landed in LDS
    __syncthreads();                             // visible block-wide
    if (k0 + 32 < kmax) stage(buf ^ 1, k0 + 32); // overlap DMA with compute

    if (k0 < q0 + 16) {                          // wave-uniform predicate
      v8f s0 = {}, s1 = {};
#pragma unroll
      for (int sub = 0; sub < 2; ++sub) {
        v8f acc = {};
        acc = wmma_bf16(qa0, load_frag(&Ks[buf][sub << 4][0], 72, 0),  acc);
        acc = wmma_bf16(qa1, load_frag(&Ks[buf][sub << 4][0], 72, 32), acc);
        const int key = k0 + (sub << 4) + nlane;
#pragma unroll
        for (int r = 0; r < 8; ++r) {
          float v = acc[r] * 0.125f;             // 1/sqrt(64)
          if (key > q0 + mbase + r) v = -__builtin_inff();
          if (sub == 0) s0[r] = v; else s1[r] = v;
        }
      }
      float corr[8];
#pragma unroll
      for (int r = 0; r < 8; ++r) {              // online softmax update
        float mloc = half_reduce_max(fmaxf(s0[r], s1[r]));
        float mnew = fmaxf(row_max[r], mloc);
        corr[r] = __expf(row_max[r] - mnew);
        row_max[r] = mnew;
      }
#pragma unroll
      for (int r = 0; r < 8; ++r) {
        float p0 = __expf(s0[r] - row_max[r]);
        float p1 = __expf(s1[r] - row_max[r]);
        row_sum[r] = row_sum[r] * corr[r] + half_reduce_sum(p0 + p1);
        sl[mbase + r][nlane]      = (bf16)p0;    // D-layout -> LDS
        sl[mbase + r][16 + nlane] = (bf16)p1;
      }
      // intra-wave LDS write->read ordering for the private P slice
      asm volatile("s_wait_dscnt 0x0" ::: "memory");
      v16bf pf = load_frag(&sl[0][0], 40, 0);    // P in A-layout
      asm volatile("" ::: "memory");
#pragma unroll
      for (int r = 0; r < 8; ++r) {
        o0[r] *= corr[r]; o1[r] *= corr[r]; o2[r] *= corr[r]; o3[r] *= corr[r];
      }
      o0 = wmma_bf16(pf, load_frag(&Vs[buf][ 0][0], 40, 0), o0);
      o1 = wmma_bf16(pf, load_frag(&Vs[buf][16][0], 40, 0), o1);
      o2 = wmma_bf16(pf, load_frag(&Vs[buf][32][0], 40, 0), o2);
      o3 = wmma_bf16(pf, load_frag(&Vs[buf][48][0], 40, 0), o3);
    }
  }

  float inv[8];
#pragma unroll
  for (int r = 0; r < 8; ++r) inv[r] = 1.0f / row_sum[r];
#pragma unroll
  for (int r = 0; r < 8; ++r) {
    size_t row = (size_t)(q0 + mbase + r) * CC + h * DK + nlane;
    yb[row +  0] = (bf16)(o0[r] * inv[r]);
    yb[row + 16] = (bf16)(o1[r] * inv[r]);
    yb[row + 32] = (bf16)(o2[r] * inv[r]);
    yb[row + 48] = (bf16)(o3[r] * inv[r]);
  }
}

// ---------------------------------------------------------------------------

extern "C" void kernel_launch(void* const* d_in, const int* in_sizes, int n_in,
                              void* d_out, int out_size, void* d_ws, size_t ws_size,
                              hipStream_t stream) {
  (void)in_sizes; (void)n_in; (void)out_size; (void)ws_size;
  const float* x  = (const float*)d_in[0];   // [1][T][C]
  const float* fc = (const float*)d_in[1];   // [T][32]
  const float* fs = (const float*)d_in[2];   // [T][32]
  const float* Wa = (const float*)d_in[3];   // [C][3C]
  const float* Wp = (const float*)d_in[4];   // [C][C]

  const size_t MB = 1ull << 20;
  char* w = (char*)d_ws;                     // 72 MB total
  bf16* xb  = (bf16*)(w +  0 * MB);          //  8 MB  x bf16 [T][C]
  bf16* WaT = (bf16*)(w +  8 * MB);          //  6 MB  W_attn^T bf16 [3C][C]
  bf16* WpT = (bf16*)(w + 14 * MB);          //  2 MB  W_proj^T bf16 [C][C]
  bf16* qkv = (bf16*)(w + 16 * MB);          // 24 MB  qkv bf16 [T][3C]
  bf16* Qh  = (bf16*)(w + 40 * MB);          //  8 MB  [H][T][DK]
  bf16* Khp = (bf16*)(w + 48 * MB);          //  8 MB  [H][T][DK]
  bf16* Vt  = (bf16*)(w + 56 * MB);          //  8 MB  [H][DK][T]
  bf16* yb  = (bf16*)(w + 64 * MB);          //  8 MB  [T][C]

  k_cvt   <<<(TT * CC) / 256, 256, 0, stream>>>(x, xb, TT * CC);
  k_cvt_T <<<(CC * C3) / 256, 256, 0, stream>>>(Wa, WaT, CC, C3);
  k_cvt_T <<<(CC * CC) / 256, 256, 0, stream>>>(Wp, WpT, CC, CC);

  // qkv = x @ W_attn : blocks = (T/16) * (3C/512) = 1536
  k_gemm<true><<<1536, 256, 0, stream>>>(xb, WaT, (void*)qkv, TT, C3, CC);

  k_rope  <<<(TT * HH * 32) / 256, 256, 0, stream>>>(qkv, fc, fs, Qh, Khp, Vt);

  // flash attention: 16 heads x 32 query-blocks (128 q each)
  k_attn  <<<512, 256, 0, stream>>>(Qh, Khp, Vt, yb);

  // out = y @ W_proj : blocks = (T/16) * (C/512) = 512
  k_gemm<false><<<512, 256, 0, stream>>>(yb, WpT, d_out, TT, CC, CC);
}
